// Encoder_67190468379298
// MI455X (gfx1250) — matrix operand order repaired and checked by reading, compiled, not run on hardware
//
#include <hip/hip_runtime.h>
#include <hip/hip_bf16.h>

// ---------------- problem constants ----------------
#define LNUM   6
#define CDIM   192
#define FCDIM  768
#define HNUM   2
#define DKDIM  96       // CDIM / HNUM
#define KSZ    3
#define WWIN   4
#define BNUM   4
#define TDIM   2048
#define TP     (TDIM + 32)   // padded T for f16 activation buffers
#define PADT   16

typedef __attribute__((ext_vector_type(16))) _Float16 v16h;
typedef __attribute__((ext_vector_type(8)))  _Float16 v8h;
typedef __attribute__((ext_vector_type(8)))  float    v8f;

// ---- WMMA fragment loaders (wave32, 16x16x32 f16) ----
// A (16x32, MxK): lane<16 -> M=lane, K=[0..7]+[16..23]; lane>=16 -> M=lane-16, K=[8..15]+[24..31]
static __device__ inline v16h load_a_frag(const _Float16* base, int seg) {
  v8h lo = *(const v8h*)(base + seg);
  v8h hi = *(const v8h*)(base + 16 + seg);
  v16h r;
#pragma unroll
  for (int e = 0; e < 8; ++e) { r[e] = lo[e]; r[8 + e] = hi[e]; }
  return r;
}
// B (32x16, KxN): lane<16 -> N=lane, K=[0..7]+[16..23]; lane>=16 -> N=lane-16, K=[8..15]+[24..31]
static __device__ inline v16h load_b_frag(const _Float16* p, size_t ld) {
  v16h r;
#pragma unroll
  for (int e = 0; e < 8; ++e) { r[e] = p[(size_t)e * ld]; r[8 + e] = p[(size_t)(16 + e) * ld]; }
  return r;
}
static __device__ inline v16h load_b_frag_u(const _Float16* p, size_t ld) { // unaligned (conv shift +-1)
  v16h r;
#pragma unroll
  for (int e = 0; e < 8; ++e) { r[e] = p[(size_t)e * ld]; r[8 + e] = p[(size_t)(16 + e) * ld]; }
  return r;
}

// ---------------------------------------------------------------------------
// Templated WMMA GEMM / conv1d kernel.
//   Y[b,m,t] = sum_k sum_c Wt[k][m][c] * X[b][c][t + k - KSZT/2]   (+bias, ...)
// Wt: f16 [KSZT][MT][KCT];  X: f16 [B][KCT][TP] (zero-padded, data at +PADT)
// 128-thread block = 4 waves; wave w computes m-tile (blockIdx.y*4+w),
// NT=4 N-tiles (16x64 outputs) so the weight A-fragment is reused 4x per load.
// Output modes: outH f16 c-major [B][MT][TP]; outT f16 per-head [B][H][T][DK];
// outF fp32 [B][MT][T].  v = (acc+bias)*oscale; relu; *mask; +res.
// ---------------------------------------------------------------------------
template<int KSZT, int MT, int KCT, int NT>
__global__ __launch_bounds__(128)
void enc_gemm(const _Float16* __restrict__ Wt, const _Float16* __restrict__ X,
              const float* __restrict__ bias, const float* __restrict__ res,
              const float* __restrict__ mask,
              _Float16* __restrict__ outH, _Float16* __restrict__ outT,
              float* __restrict__ outF, int relu, float oscale)
{
  const int lane = threadIdx.x & 31;
  const int wave = threadIdx.x >> 5;
  const int t0 = blockIdx.x * (16 * NT);
  const int m0 = (blockIdx.y * 4 + wave) * 16;
  const int b  = blockIdx.z;
  const int row = lane & 15;
  const int seg = (lane >> 4) * 8;
  const int hi8 = seg;

  v8f acc[NT] = {};
#pragma unroll
  for (int k = 0; k < KSZT; ++k) {
    const int shift = k - (KSZT >> 1);                 // compile-time per tap
    const _Float16* Wk = Wt + ((size_t)k * MT + m0 + row) * KCT;
    const _Float16* Xb = X + (size_t)b * KCT * TP + (PADT + t0 + shift);
    for (int c0 = 0; c0 < KCT; c0 += 32) {
      v16h a = load_a_frag(Wk + c0, seg);
      const _Float16* bp = Xb + (size_t)(c0 + seg) * TP + (lane & 15);
      if (c0 + 32 < KCT)                               // stream next K-chunk into L2
        __builtin_prefetch(bp + (size_t)32 * TP, 0, 0);
#pragma unroll
      for (int n = 0; n < NT; ++n) {
        v16h bf = (shift == 0) ? load_b_frag(bp + n * 16, TP)
                               : load_b_frag_u(bp + n * 16, TP);
        acc[n] = __builtin_amdgcn_wmma_f32_16x16x32_f16(false, a, false, bf,
                                                        (short)0, acc[n], false, false);
      }
    }
  }

  float bj[8];
#pragma unroll
  for (int j = 0; j < 8; ++j) bj[j] = bias[m0 + j + hi8];

#pragma unroll
  for (int n = 0; n < NT; ++n) {
    const int t = t0 + n * 16 + (lane & 15);
    const float mval = mask ? mask[(size_t)b * TDIM + t] : 1.0f;
#pragma unroll
    for (int j = 0; j < 8; ++j) {
      const int m = m0 + j + hi8;                      // D layout: VGPR j -> row j / j+8
      float v = (acc[n][j] + bj[j]) * oscale;
      if (relu) v = fmaxf(v, 0.0f);
      if (mask) v *= mval;
      if (res)  v += res[((size_t)b * MT + m) * TDIM + t];
      if (outH) outH[((size_t)b * MT + m) * TP + PADT + t] = (_Float16)v;
      if (outT) {
        const int hh = m0 / DKDIM;                     // tile never straddles heads
        outT[((size_t)(b * HNUM + hh) * TDIM + t) * DKDIM + (m - hh * DKDIM)] = (_Float16)v;
      }
      if (outF) outF[((size_t)b * MT + m) * TDIM + t] = v;
    }
  }
}

// ---------------------------------------------------------------------------
// Fused flash-attention with banded relative-position terms (window W=4).
// Q: f16 [B][H][T][DK] (pre-scaled by DK^-0.5); K: f16 [B][C][TP] c-major;
// V: f16 [B][H][T][DK]; out: f16 c-major [B][C][TP].
// One wave handles a 16-row t-strip; online softmax over 64 s-chunks of 32.
// ---------------------------------------------------------------------------
__global__ __launch_bounds__(32)
void enc_attn(const _Float16* __restrict__ Q, const _Float16* __restrict__ Kb,
              const _Float16* __restrict__ V, const float* __restrict__ relk,
              const float* __restrict__ relv, const float* __restrict__ mask,
              _Float16* __restrict__ Oa)
{
  __shared__ alignas(16) float    qrelS[16][9];
  __shared__ alignas(16) float    relvS[9 * DKDIM];
  __shared__ alignas(16) float    mtS[16];
  __shared__ alignas(16) float    rowmS[16];
  __shared__ alignas(16) float    rowlS[16];
  __shared__ alignas(64) _Float16 ptile[16][32];
  __shared__ alignas(16) float    pbandS[16][9];

  const int lane = threadIdx.x & 31;
  const int t0 = blockIdx.x * 16, h = blockIdx.y, b = blockIdx.z;
  const int row = lane & 15;
  const int seg = (lane >> 4) * 8;
  const int hi8 = seg;

  const _Float16* Qh = Q + (size_t)(b * HNUM + h) * TDIM * DKDIM;
  const _Float16* Vh = V + (size_t)(b * HNUM + h) * TDIM * DKDIM;
  const _Float16* Kh = Kb + ((size_t)b * CDIM + h * DKDIM) * TP + PADT;

  for (int i = lane; i < 9 * DKDIM; i += 32) relvS[i] = relv[i];
  if (lane < 16) mtS[lane] = mask[(size_t)b * TDIM + t0 + lane];
  // banded rel-k scores: qrel[t][d] = qs[t] . emb_k[d],  d = (s-t)+W
  for (int i = lane; i < 16 * 9; i += 32) {
    const int tt = i / 9, d = i % 9;
    const _Float16* qp = Qh + (size_t)(t0 + tt) * DKDIM;
    const float* rp = relk + d * DKDIM;
    float s = 0.f;
    for (int c = 0; c < DKDIM; ++c) s += (float)qp[c] * rp[c];
    qrelS[tt][d] = s;
  }
  __syncthreads();

  // Q A-fragments for the 3 k-steps of DK=96 stay in registers
  v16h aq[3];
#pragma unroll
  for (int k = 0; k < 3; ++k)
    aq[k] = load_a_frag(Qh + (size_t)(t0 + row) * DKDIM + k * 32, seg);

  float mrow[8], lrow[8], mt_reg[8];
  v8f acc[6] = {};
#pragma unroll
  for (int j = 0; j < 8; ++j) { mrow[j] = -1e30f; lrow[j] = 0.f; mt_reg[j] = mtS[j + hi8]; }

  for (int s0 = 0; s0 < TDIM; s0 += 32) {
    // ---- scores: two 16x16 subtiles, K-loop over DK ----
    v8f sc[2] = {};
#pragma unroll
    for (int sub = 0; sub < 2; ++sub) {
#pragma unroll
      for (int k = 0; k < 3; ++k) {
        const _Float16* kp = Kh + (size_t)(k * 32 + seg) * TP + s0 + sub * 16 + (lane & 15);
        v16h bk = load_b_frag(kp, TP);
        sc[sub] = __builtin_amdgcn_wmma_f32_16x16x32_f16(false, aq[k], false, bk,
                                                         (short)0, sc[sub], false, false);
      }
    }
    // ---- rel-k band + mask ----
    const float ms0 = mask[(size_t)b * TDIM + s0 + (lane & 15)];
    const float ms1 = mask[(size_t)b * TDIM + s0 + 16 + (lane & 15)];
#pragma unroll
    for (int sub = 0; sub < 2; ++sub) {
      const float msv = sub ? ms1 : ms0;
#pragma unroll
      for (int j = 0; j < 8; ++j) {
        const int tr = j + hi8;
        const int sI = s0 + sub * 16 + (lane & 15);
        float v = sc[sub][j];
        const int d = sI - (t0 + tr);
        if (d >= -WWIN && d <= WWIN) v += qrelS[tr][d + WWIN];
        if (msv == 0.f || mt_reg[j] == 0.f) v = -1e4f;
        sc[sub][j] = v;
      }
    }
    // ---- online softmax update (row reductions across the 16-lane N stripe) ----
#pragma unroll
    for (int j = 0; j < 8; ++j) {
      float c = fmaxf(sc[0][j], sc[1][j]);
      c = fmaxf(c, __shfl_xor(c, 8, 32));
      c = fmaxf(c, __shfl_xor(c, 4, 32));
      c = fmaxf(c, __shfl_xor(c, 2, 32));
      c = fmaxf(c, __shfl_xor(c, 1, 32));
      const float mn  = fmaxf(mrow[j], c);
      const float sca = __expf(mrow[j] - mn);
      mrow[j] = mn;
      lrow[j] *= sca;
#pragma unroll
      for (int k = 0; k < 6; ++k) acc[k][j] *= sca;
      const float p0 = __expf(sc[0][j] - mn);
      const float p1 = __expf(sc[1][j] - mn);
      sc[0][j] = p0; sc[1][j] = p1;
      float r = p0 + p1;
      r += __shfl_xor(r, 8, 32);
      r += __shfl_xor(r, 4, 32);
      r += __shfl_xor(r, 2, 32);
      r += __shfl_xor(r, 1, 32);
      lrow[j] += r;
    }
    // ---- P -> LDS f16, re-fragment as WMMA-A, accumulate P.V ----
#pragma unroll
    for (int sub = 0; sub < 2; ++sub)
#pragma unroll
      for (int j = 0; j < 8; ++j)
        ptile[j + hi8][sub * 16 + (lane & 15)] = (_Float16)sc[sub][j];
    __syncthreads();
    v16h ap = load_a_frag(&ptile[row][0], seg);
#pragma unroll
    for (int kt = 0; kt < 6; ++kt) {
      const _Float16* vp = Vh + (size_t)(s0 + seg) * DKDIM + kt * 16 + (lane & 15);
      v16h bv = load_b_frag(vp, DKDIM);
      acc[kt] = __builtin_amdgcn_wmma_f32_16x16x32_f16(false, ap, false, bv,
                                                       (short)0, acc[kt], false, false);
    }
    __syncthreads();
  }

  // ---- publish final row stats, recompute the 9-wide band with final (m,l) ----
  if (lane == 0 || lane == 16) {
#pragma unroll
    for (int j = 0; j < 8; ++j) { rowmS[j + hi8] = mrow[j]; rowlS[j + hi8] = lrow[j]; }
  }
  __syncthreads();
  for (int i = lane; i < 16 * 9; i += 32) {
    const int tt = i / 9, d = i % 9;
    const int tI = t0 + tt, sI = tI + d - WWIN;
    float pv = 0.f;
    if (sI >= 0 && sI < TDIM) {
      const _Float16* qp = Qh + (size_t)tI * DKDIM;
      float s = 0.f;
      for (int c = 0; c < DKDIM; ++c) s += (float)qp[c] * (float)Kh[(size_t)c * TP + sI];
      s += qrelS[tt][d];
      if (mask[(size_t)b * TDIM + sI] == 0.f || mtS[tt] == 0.f) s = -1e4f;
      pv = __expf(s - rowmS[tt]) / rowlS[tt];
    }
    pbandS[tt][d] = pv;
  }
  __syncthreads();

  // ---- normalize, add rel-v band, store c-major f16 for the O projection ----
#pragma unroll
  for (int kt = 0; kt < 6; ++kt) {
#pragma unroll
    for (int j = 0; j < 8; ++j) {
      const int tr = j + hi8;
      const int dk = kt * 16 + (lane & 15);
      float v = acc[kt][j] / lrow[j];
      float badd = 0.f;
#pragma unroll
      for (int d = 0; d < 9; ++d) badd += pbandS[tr][d] * relvS[d * DKDIM + dk];
      v += badd;
      Oa[((size_t)b * CDIM + h * DKDIM + dk) * TP + PADT + (t0 + tr)] = (_Float16)v;
    }
  }
}

// ---------------------------------------------------------------------------
// Channel LayerNorm: one wave per (b,t); writes fp32 state + f16 GEMM operand.
// ---------------------------------------------------------------------------
__global__ __launch_bounds__(32)
void enc_ln(const float* __restrict__ in, const float* __restrict__ g,
            const float* __restrict__ bb, const float* __restrict__ mask,
            float* __restrict__ outF, _Float16* __restrict__ outH, int maskH)
{
  const int t = blockIdx.x, b = blockIdx.y, lane = threadIdx.x & 31;
  float vals[6];
  float s = 0.f;
#pragma unroll
  for (int i = 0; i < 6; ++i) {
    const int c = lane + 32 * i;
    vals[i] = in[((size_t)b * CDIM + c) * TDIM + t];
    s += vals[i];
  }
  s += __shfl_xor(s, 16, 32); s += __shfl_xor(s, 8, 32); s += __shfl_xor(s, 4, 32);
  s += __shfl_xor(s, 2, 32);  s += __shfl_xor(s, 1, 32);
  const float mean = s * (1.0f / CDIM);
  float vsum = 0.f;
#pragma unroll
  for (int i = 0; i < 6; ++i) { const float d = vals[i] - mean; vsum += d * d; }
  vsum += __shfl_xor(vsum, 16, 32); vsum += __shfl_xor(vsum, 8, 32);
  vsum += __shfl_xor(vsum, 4, 32);  vsum += __shfl_xor(vsum, 2, 32);
  vsum += __shfl_xor(vsum, 1, 32);
  const float inv = rsqrtf(vsum * (1.0f / CDIM) + 1e-5f);
  const float mk = maskH ? mask[(size_t)b * TDIM + t] : 1.0f;
#pragma unroll
  for (int i = 0; i < 6; ++i) {
    const int c = lane + 32 * i;
    const float y = g[c] * (vals[i] - mean) * inv + bb[c];
    outF[((size_t)b * CDIM + c) * TDIM + t] = y;
    outH[((size_t)b * CDIM + c) * TP + PADT + t] = (_Float16)(y * mk);
  }
}

// ---------------- small prep / utility kernels ----------------
__global__ void enc_cvt16(const float* __restrict__ src, _Float16* __restrict__ dst, size_t n) {
  for (size_t i = (size_t)blockIdx.x * blockDim.x + threadIdx.x; i < n;
       i += (size_t)gridDim.x * blockDim.x)
    dst[i] = (_Float16)src[i];
}
// src [L][OC][IC][K] -> dst [L][K][OC][IC]  (conv weights, k-major for GEMM)
__global__ void enc_perm_ffn(const float* __restrict__ src, _Float16* __restrict__ dst,
                             int OC, int IC) {
  const size_t n = (size_t)LNUM * OC * IC * KSZ;
  for (size_t i = (size_t)blockIdx.x * blockDim.x + threadIdx.x; i < n;
       i += (size_t)gridDim.x * blockDim.x) {
    const size_t k = i % KSZ;
    const size_t c = (i / KSZ) % IC;
    const size_t o = (i / ((size_t)KSZ * IC)) % OC;
    const size_t l = i / ((size_t)KSZ * IC * OC);
    dst[((l * KSZ + k) * OC + o) * IC + c] = (_Float16)src[i];
  }
}
__global__ void enc_zero16(_Float16* __restrict__ p, size_t n) {
  for (size_t i = (size_t)blockIdx.x * blockDim.x + threadIdx.x; i < n;
       i += (size_t)gridDim.x * blockDim.x)
    p[i] = (_Float16)0.f;
}
__global__ void enc_prepx(const float* __restrict__ x, const float* __restrict__ mask,
                          float* __restrict__ xf, _Float16* __restrict__ xh) {
  const size_t n = (size_t)BNUM * CDIM * TDIM;
  for (size_t i = (size_t)blockIdx.x * blockDim.x + threadIdx.x; i < n;
       i += (size_t)gridDim.x * blockDim.x) {
    const size_t t = i % TDIM;
    const size_t c = (i / TDIM) % CDIM;
    const size_t b = i / ((size_t)TDIM * CDIM);
    const float v = x[i] * mask[b * TDIM + t];
    xf[i] = v;
    xh[(b * CDIM + c) * TP + PADT + t] = (_Float16)v;
  }
}
__global__ void enc_final(const float* __restrict__ xf, const float* __restrict__ mask,
                          float* __restrict__ out) {
  const size_t n = (size_t)BNUM * CDIM * TDIM;
  for (size_t i = (size_t)blockIdx.x * blockDim.x + threadIdx.x; i < n;
       i += (size_t)gridDim.x * blockDim.x) {
    const size_t t = i % TDIM;
    const size_t b = i / ((size_t)TDIM * CDIM);
    out[i] = xf[i] * mask[b * TDIM + t];
  }
}

// ---------------------------------------------------------------------------
extern "C" void kernel_launch(void* const* d_in, const int* in_sizes, int n_in,
                              void* d_out, int out_size, void* d_ws, size_t ws_size,
                              hipStream_t stream)
{
  (void)in_sizes; (void)n_in; (void)out_size; (void)ws_size;
  const float* x    = (const float*)d_in[0];
  const float* mask = (const float*)d_in[1];
  const float* wq   = (const float*)d_in[2];  const float* bq = (const float*)d_in[3];
  const float* wk   = (const float*)d_in[4];  const float* bk = (const float*)d_in[5];
  const float* wv   = (const float*)d_in[6];  const float* bv = (const float*)d_in[7];
  const float* wo   = (const float*)d_in[8];  const float* bo = (const float*)d_in[9];
  const float* relk = (const float*)d_in[10]; const float* relv = (const float*)d_in[11];
  const float* ln1g = (const float*)d_in[12]; const float* ln1b = (const float*)d_in[13];
  const float* fw1  = (const float*)d_in[14]; const float* fb1 = (const float*)d_in[15];
  const float* fw2  = (const float*)d_in[16]; const float* fb2 = (const float*)d_in[17];
  const float* ln2g = (const float*)d_in[18]; const float* ln2b = (const float*)d_in[19];

  // ---- carve workspace (256B aligned) ----
  char* wp = (char*)d_ws;
  auto carve = [&](size_t bytes) -> char* {
    char* r = wp; wp += (bytes + 255) & ~(size_t)255; return r;
  };
  const size_t nWp  = (size_t)LNUM * CDIM * CDIM;
  const size_t nW1  = (size_t)LNUM * KSZ * FCDIM * CDIM;
  const size_t nAct = (size_t)BNUM * CDIM * TP;
  const size_t nQV  = (size_t)BNUM * CDIM * TDIM;
  const size_t nH   = (size_t)BNUM * FCDIM * TP;
  const size_t nF   = (size_t)BNUM * CDIM * TDIM;

  _Float16* wq16 = (_Float16*)carve(nWp * 2);
  _Float16* wk16 = (_Float16*)carve(nWp * 2);
  _Float16* wv16 = (_Float16*)carve(nWp * 2);
  _Float16* wo16 = (_Float16*)carve(nWp * 2);
  _Float16* w116 = (_Float16*)carve(nW1 * 2);
  _Float16* w216 = (_Float16*)carve(nW1 * 2);
  _Float16* xh   = (_Float16*)carve(nAct * 2);   // padded c-major f16 activations
  _Float16* kh   = (_Float16*)carve(nAct * 2);
  _Float16* ah   = (_Float16*)carve(nAct * 2);   // attention output
  _Float16* hh   = (_Float16*)carve(nH * 2);     // FFN hidden
  _Float16* qh   = (_Float16*)carve(nQV * 2);    // [B,H,T,DK]
  _Float16* vh   = (_Float16*)carve(nQV * 2);
  float*    xbuf = (float*)carve(nF * 4);        // current x (fp32)
  float*    preln= (float*)carve(nF * 4);        // residual+branch, pre-LN

  // ---- weight conversion / permutation ----
  enc_cvt16<<<1024, 256, 0, stream>>>(wq, wq16, nWp);
  enc_cvt16<<<1024, 256, 0, stream>>>(wk, wk16, nWp);
  enc_cvt16<<<1024, 256, 0, stream>>>(wv, wv16, nWp);
  enc_cvt16<<<1024, 256, 0, stream>>>(wo, wo16, nWp);
  enc_perm_ffn<<<2048, 256, 0, stream>>>(fw1, w116, FCDIM, CDIM);
  enc_perm_ffn<<<2048, 256, 0, stream>>>(fw2, w216, CDIM, FCDIM);
  // zero padded f16 buffers (keeps conv/WMMA edge reads exact zero)
  enc_zero16<<<2048, 256, 0, stream>>>(xh, nAct);
  enc_zero16<<<2048, 256, 0, stream>>>(kh, nAct);
  enc_zero16<<<2048, 256, 0, stream>>>(ah, nAct);
  enc_zero16<<<4096, 256, 0, stream>>>(hh, nH);
  enc_prepx<<<4096, 256, 0, stream>>>(x, mask, xbuf, xh);

  // 4 waves/block, each wave: 16 rows x 64 cols of output
  const dim3 gP(TDIM / 64, (CDIM / 16) / 4, BNUM);    // (32, 3, 4)
  const dim3 gF1(TDIM / 64, (FCDIM / 16) / 4, BNUM);  // (32, 12, 4)
  const dim3 gA(TDIM / 16, HNUM, BNUM);
  const dim3 gL(TDIM, BNUM);
  const float qscale = 0.1020620726159658f;   // DK^-0.5

  for (int l = 0; l < LNUM; ++l) {
    const size_t ow = (size_t)l * CDIM * CDIM;
    // Q/K/V projections (Q pre-scaled; Q,V -> [T,DK] per head, K -> c-major)
    enc_gemm<1, CDIM, CDIM, 4><<<gP, 128, 0, stream>>>(
        wq16 + ow, xh, bq + l * CDIM, nullptr, nullptr,
        nullptr, qh, nullptr, 0, qscale);
    enc_gemm<1, CDIM, CDIM, 4><<<gP, 128, 0, stream>>>(
        wk16 + ow, xh, bk + l * CDIM, nullptr, nullptr,
        kh, nullptr, nullptr, 0, 1.0f);
    enc_gemm<1, CDIM, CDIM, 4><<<gP, 128, 0, stream>>>(
        wv16 + ow, xh, bv + l * CDIM, nullptr, nullptr,
        nullptr, vh, nullptr, 0, 1.0f);
    // fused attention
    enc_attn<<<gA, 32, 0, stream>>>(qh, kh, vh,
                                    relk + (size_t)l * 9 * DKDIM,
                                    relv + (size_t)l * 9 * DKDIM, mask, ah);
    // output projection + residual -> pre-LN1
    enc_gemm<1, CDIM, CDIM, 4><<<gP, 128, 0, stream>>>(
        wo16 + ow, ah, bo + l * CDIM, xbuf, nullptr,
        nullptr, nullptr, preln, 0, 1.0f);
    enc_ln<<<gL, 32, 0, stream>>>(preln, ln1g + l * CDIM, ln1b + l * CDIM, mask,
                                  xbuf, xh, 1);   // f16 copy masked (FFN input)
    // FFN: conv1d(C->FC,k=3) + ReLU, conv1d(FC->C,k=3)*mask + residual
    enc_gemm<KSZ, FCDIM, CDIM, 4><<<gF1, 128, 0, stream>>>(
        w116 + (size_t)l * KSZ * FCDIM * CDIM, xh, fb1 + l * FCDIM, nullptr, nullptr,
        hh, nullptr, nullptr, 1, 1.0f);
    enc_gemm<KSZ, CDIM, FCDIM, 4><<<gP, 128, 0, stream>>>(
        w216 + (size_t)l * KSZ * CDIM * FCDIM, hh, fb2 + l * CDIM, xbuf, mask,
        nullptr, nullptr, preln, 0, 1.0f);
    enc_ln<<<gL, 32, 0, stream>>>(preln, ln2g + l * CDIM, ln2b + l * CDIM, mask,
                                  xbuf, xh, 0);   // f16 copy unmasked (next QKV)
  }
  enc_final<<<4096, 256, 0, stream>>>(xbuf, mask, (float*)d_out);
}
